// acLSTM_31533649887640
// MI455X (gfx1250) — matrix operand order repaired and checked by reading, not compiled, not run on hardware
//
#include <hip/hip_runtime.h>
#include <hip/hip_bf16.h>

// ---------------------------------------------------------------------------
// 3-layer LSTM + autoregressive decode, persistent-kernel, bf16 WMMA (gfx1250)
// B=32, H=1024, IN=OUT=171, T = 128 seed + 384 gen = 512 serial steps.
// Round 3 changes:
//  - launder a zero *offset* (not the pointers) per step: weight loads remain
//    global_load (SADDR form, single counter) instead of flat_load, while
//    still preventing LICM from hoisting+spilling the weight fragments.
//  - __restrict__ restored on weight pointers.
// ---------------------------------------------------------------------------

typedef __attribute__((ext_vector_type(16))) __bf16 v16bf;
typedef __attribute__((ext_vector_type(8)))  float  v8f;

#define B_     32
#define SEEDL  128
#define TTOT   512
#define IN_    171
#define INP    192      // IN padded to multiple of 32 (K of layer-1 x-GEMM)
#define H_     1024
#define G4H    4096
#define OUT_   171
#define OUTP   176      // OUT padded to multiple of 16 (N of decoder GEMM)
#define NWG    64
#define NTHR   256
#define NT_ALL (NWG * NTHR)
#define LDSW   1024     // LDS row width (elements) for staged A matrix

// ------------------------------ small utility kernels ----------------------

__global__ void zero_ws_kernel(unsigned* p, size_t nwords) {
    size_t i = (size_t)blockIdx.x * blockDim.x + threadIdx.x;
    if (i < nwords) p[i] = 0u;
}

// fp32 -> bf16 with zero padding of rows/cols (row-major [drows x dcols])
__global__ void cvt_pad_kernel(const float* __restrict__ src, __bf16* __restrict__ dst,
                               int drows, int dcols, int srows, int scols) {
    int i = blockIdx.x * blockDim.x + threadIdx.x;
    int total = drows * dcols;
    if (i >= total) return;
    int r = i / dcols, c = i % dcols;
    float v = (r < srows && c < scols) ? src[(size_t)r * scols + c] : 0.0f;
    dst[i] = (__bf16)v;
}

// ------------------------------ device helpers -----------------------------

// Cooperative copy of a contiguous [32 x K] bf16 matrix into LDS with a
// per-row rotate swizzle: element (m,k) -> lds[m*LDSW + ((k + 8*m) & (LDSW-1))].
// Row m shifts 4 dwords -> fragment rows land on distinct banks.
__device__ inline void stage_a_lds(__bf16* lds, const __bf16* __restrict__ src, int K) {
    int nchunks = (B_ * K) >> 3;                 // 16-byte chunks
    for (int c = threadIdx.x; c < nchunks; c += NTHR) {
        int e = c << 3;
        int m = e / K, k = e % K;
        int d = m * LDSW + ((k + 8 * m) & (LDSW - 1));
        *(uint4*)(lds + d) = *(const uint4*)(src + e);
    }
}

// A fragment from swizzled LDS: 16x32 bf16. lane L: M = m0 + L%16,
// elements 0..7 = K kb..kb+7, elements 8..15 = K kb+16..kb+23, kb = k0+8*(L>=16)
__device__ inline v16bf lds_a_frag(const __bf16* lds, int m0, int k0, int lane) {
    int m  = m0 + (lane & 15);
    int kb = k0 + ((lane >> 4) << 3);
    int i0 = (kb + 8 * m) & (LDSW - 1);
    int i1 = (kb + 16 + 8 * m) & (LDSW - 1);
    const __bf16* row = lds + m * LDSW;
    union { v16bf v; uint4 q[2]; } u;
    u.q[0] = *(const uint4*)(row + i0);
    u.q[1] = *(const uint4*)(row + i1);
    return u.v;
}

// B fragment: 32x16 bf16 of W^T => W[n][k] contiguous in k.
// lane L: N = n0 + L%16, elements 0..7 = K kb..kb+7, 8..15 = K kb+16..kb+23, kb = k0+8*(L>=16)
__device__ inline v16bf load_b_frag(const __bf16* __restrict__ W, int ldk,
                                    int n0, int k0, int lane) {
    int n  = n0 + (lane & 15);
    int kb = k0 + ((lane >> 4) << 3);
    const __bf16* p = W + (size_t)n * ldk + kb;
    union { v16bf v; uint4 q[2]; } u;
    u.q[0] = *(const uint4*)(p);
    u.q[1] = *(const uint4*)(p + 16);
    return u.v;
}

// GEMM accumulate: A from swizzled LDS, B (weights) from global/L2.
__device__ inline v8f mm_acc_lds(v8f acc, const __bf16* lds,
                                 const __bf16* __restrict__ W, int ldk,
                                 int Ktot, int m0, int n0, int lane) {
#pragma unroll 4
    for (int k0 = 0; k0 < Ktot; k0 += 32) {
        v16bf a = lds_a_frag(lds, m0, k0, lane);
        v16bf b = load_b_frag(W, ldk, n0, k0, lane);
        acc = __builtin_amdgcn_wmma_f32_16x16x32_bf16(
            false, a, false, b, (short)0, acc, false, false);
    }
    return acc;
}

// GEMM accumulate with A from global (used only by the tiny decoder GEMM).
__device__ inline v8f mm_acc_glb(v8f acc, const __bf16* __restrict__ A, int lda,
                                 const __bf16* __restrict__ W, int ldk,
                                 int Ktot, int m0, int n0, int lane) {
#pragma unroll 4
    for (int k0 = 0; k0 < Ktot; k0 += 32) {
        int m  = m0 + (lane & 15);
        int kb = k0 + ((lane >> 4) << 3);
        const __bf16* p = A + (size_t)m * lda + kb;
        union { v16bf v; uint4 q[2]; } ua;
        ua.q[0] = *(const uint4*)(p);
        ua.q[1] = *(const uint4*)(p + 16);
        v16bf b = load_b_frag(W, ldk, n0, k0, lane);
        acc = __builtin_amdgcn_wmma_f32_16x16x32_bf16(
            false, ua.v, false, b, (short)0, acc, false, false);
    }
    return acc;
}

// C/D layout: lane L holds column n = n0+L%16, rows m = m0 + (L<16?0:8) + r
__device__ inline void store_tile(float* dst, int ldn, int m0, int n0, int lane, v8f acc) {
    int n     = n0 + (lane & 15);
    int mbase = m0 + ((lane >> 4) << 3);
#pragma unroll
    for (int r = 0; r < 8; ++r)
        dst[(size_t)(mbase + r) * ldn + n] = acc[r];
}

// grid-wide barrier: monotonic ticket counter (reset to 0 by zero_ws each launch)
__device__ inline void grid_bar(unsigned* bar) {
    __syncthreads();
    if (threadIdx.x == 0) {
        __threadfence();
        unsigned t = atomicAdd(bar, 1u);
        unsigned target = (t / NWG + 1u) * (unsigned)NWG;
        while (__atomic_load_n(bar, __ATOMIC_RELAXED) < target)
            __builtin_amdgcn_s_sleep(2);
        __threadfence();
    }
    __syncthreads();
}

__device__ inline float sigm(float x) { return 1.0f / (1.0f + __expf(-x)); }

__device__ inline void act_phase(const float* __restrict__ gates,
                                 const float* __restrict__ bih,
                                 const float* __restrict__ bhh,
                                 float* __restrict__ c, __bf16* __restrict__ h, int gt) {
    for (int idx = gt; idx < B_ * H_; idx += NT_ALL) {
        int b = idx >> 10, j = idx & (H_ - 1);
        const float* gb = gates + (size_t)b * G4H;
        float gi = gb[j]          + bih[j]          + bhh[j];
        float gf = gb[j + H_]     + bih[j + H_]     + bhh[j + H_];
        float gg = gb[j + 2*H_]   + bih[j + 2*H_]   + bhh[j + 2*H_];
        float go = gb[j + 3*H_]   + bih[j + 3*H_]   + bhh[j + 3*H_];
        float cn = sigm(gf) * c[idx] + sigm(gi) * tanhf(gg);
        float hn = sigm(go) * tanhf(cn);
        c[idx] = cn;
        h[idx] = (__bf16)hn;
    }
}

// ------------------------------ persistent kernel --------------------------

__global__ void __launch_bounds__(NTHR, 1)
lstm_persistent(const float* __restrict__ seed,
                const __bf16* __restrict__ Wih1_, const __bf16* __restrict__ Whh1_,
                const float* __restrict__ bih1,   const float* __restrict__ bhh1,
                const __bf16* __restrict__ Wih2_, const __bf16* __restrict__ Whh2_,
                const float* __restrict__ bih2,   const float* __restrict__ bhh2,
                const __bf16* __restrict__ Wih3_, const __bf16* __restrict__ Whh3_,
                const float* __restrict__ bih3,   const float* __restrict__ bhh3,
                const __bf16* __restrict__ Wdec_, const float* __restrict__ bdec,
                __bf16* xbuf, __bf16* h1, __bf16* h2, __bf16* h3,
                float* c1, float* c2, float* c3,
                float* gates, float* out, unsigned* bar)
{
    __shared__ __bf16 smem[B_ * LDSW];          // 64 KB staged A matrix

    const int lane  = threadIdx.x & 31;
    const int wave  = blockIdx.x * (NTHR / 32) + (threadIdx.x >> 5); // 0..511
    const int gt    = blockIdx.x * NTHR + threadIdx.x;
    const int m0    = (wave & 1) * 16;          // 2 M-tiles (B=32)
    const int n0    = (wave >> 1) * 16;         // 256 N-tiles (4H=4096)

    for (int t = 0; t < TTOT; ++t) {
        // Launder a zero element-offset each step: weight addresses become
        // non-loop-invariant (no LICM hoist + scratch spill of fragments),
        // but pointer provenance is preserved so loads stay global_load
        // (SADDR form) rather than degrading to flat_load.
        size_t lz = 0;
        asm volatile("" : "+s"(lz));
        const __bf16* __restrict__ Wih1 = Wih1_ + lz;
        const __bf16* __restrict__ Whh1 = Whh1_ + lz;
        const __bf16* __restrict__ Wih2 = Wih2_ + lz;
        const __bf16* __restrict__ Whh2 = Whh2_ + lz;
        const __bf16* __restrict__ Wih3 = Wih3_ + lz;
        const __bf16* __restrict__ Whh3 = Whh3_ + lz;
        const __bf16* __restrict__ Wdec = Wdec_ + lz;

        // ---- stage x (seed phase converts fp32 seed; gen phase: decoder wrote it)
        if (t < SEEDL) {
            for (int idx = gt; idx < B_ * INP; idx += NT_ALL) {
                int b = idx / INP, i = idx % INP;
                float v = (i < IN_) ? seed[((size_t)b * SEEDL + t) * IN_ + i] : 0.0f;
                xbuf[idx] = (__bf16)v;
            }
        }
        grid_bar(bar);

        // ---- layer 1: gates = x@Wih1^T + h1@Whh1^T
        {
            v8f acc = {};
            stage_a_lds(smem, xbuf, INP);  __syncthreads();
            acc = mm_acc_lds(acc, smem, Wih1, INP, INP, m0, n0, lane);
            __syncthreads();
            stage_a_lds(smem, h1, H_);     __syncthreads();
            acc = mm_acc_lds(acc, smem, Whh1, H_, H_, m0, n0, lane);
            store_tile(gates, G4H, m0, n0, lane, acc);
        }
        grid_bar(bar);
        act_phase(gates, bih1, bhh1, c1, h1, gt);
        grid_bar(bar);

        // ---- layer 2: gates = h1@Wih2^T + h2@Whh2^T
        {
            v8f acc = {};
            stage_a_lds(smem, h1, H_);     __syncthreads();
            acc = mm_acc_lds(acc, smem, Wih2, H_, H_, m0, n0, lane);
            __syncthreads();
            stage_a_lds(smem, h2, H_);     __syncthreads();
            acc = mm_acc_lds(acc, smem, Whh2, H_, H_, m0, n0, lane);
            store_tile(gates, G4H, m0, n0, lane, acc);
        }
        grid_bar(bar);
        act_phase(gates, bih2, bhh2, c2, h2, gt);
        grid_bar(bar);

        // ---- layer 3: gates = h2@Wih3^T + h3@Whh3^T
        {
            v8f acc = {};
            stage_a_lds(smem, h2, H_);     __syncthreads();
            acc = mm_acc_lds(acc, smem, Wih3, H_, H_, m0, n0, lane);
            __syncthreads();
            stage_a_lds(smem, h3, H_);     __syncthreads();
            acc = mm_acc_lds(acc, smem, Whh3, H_, H_, m0, n0, lane);
            store_tile(gates, G4H, m0, n0, lane, acc);
        }
        grid_bar(bar);
        act_phase(gates, bih3, bhh3, c3, h3, gt);
        grid_bar(bar);

        // ---- decoder: out = h3 @ Wdec^T + bdec ; feed back into xbuf
        if (wave < 2 * (OUTP / 16)) {           // 22 waves cover 2 x 11 tiles
            int dn0 = (wave >> 1) * 16;
            v8f acc = {};
            acc = mm_acc_glb(acc, h3, H_, Wdec, H_, H_, m0, dn0, lane);
            int n     = dn0 + (lane & 15);
            int mbase = m0  + ((lane >> 4) << 3);
            if (n < OUT_) {
                float bv = bdec[n];
#pragma unroll
                for (int r = 0; r < 8; ++r) {
                    float v = acc[r] + bv;
                    int b = mbase + r;
                    out[((size_t)b * TTOT + t) * OUT_ + n] = v;
                    xbuf[b * INP + n] = (__bf16)v;   // next-step input
                }
            }
        }
        grid_bar(bar);                           // xbuf/gates safe for next step
    }
}

// ------------------------------ host entry --------------------------------

extern "C" void kernel_launch(void* const* d_in, const int* in_sizes, int n_in,
                              void* d_out, int out_size, void* d_ws, size_t ws_size,
                              hipStream_t stream) {
    const float* seed  = (const float*)d_in[0];
    const float* fWih1 = (const float*)d_in[1];
    const float* fWhh1 = (const float*)d_in[2];
    const float* bih1  = (const float*)d_in[3];
    const float* bhh1  = (const float*)d_in[4];
    const float* fWih2 = (const float*)d_in[5];
    const float* fWhh2 = (const float*)d_in[6];
    const float* bih2  = (const float*)d_in[7];
    const float* bhh2  = (const float*)d_in[8];
    const float* fWih3 = (const float*)d_in[9];
    const float* fWhh3 = (const float*)d_in[10];
    const float* bih3  = (const float*)d_in[11];
    const float* bhh3  = (const float*)d_in[12];
    const float* fWdec = (const float*)d_in[13];
    const float* bdec  = (const float*)d_in[14];
    float* out = (float*)d_out;

    char* ws = (char*)d_ws;
    size_t off = 0;
    auto alloc = [&](size_t bytes) -> char* {
        char* p = ws + off;
        off += (bytes + 255) & ~(size_t)255;
        return p;
    };

    __bf16* wWih1 = (__bf16*)alloc((size_t)G4H * INP * 2);
    __bf16* wWhh1 = (__bf16*)alloc((size_t)G4H * H_  * 2);
    __bf16* wWih2 = (__bf16*)alloc((size_t)G4H * H_  * 2);
    __bf16* wWhh2 = (__bf16*)alloc((size_t)G4H * H_  * 2);
    __bf16* wWih3 = (__bf16*)alloc((size_t)G4H * H_  * 2);
    __bf16* wWhh3 = (__bf16*)alloc((size_t)G4H * H_  * 2);
    __bf16* wWdec = (__bf16*)alloc((size_t)OUTP * H_ * 2);

    size_t state_begin = off;
    __bf16*   xbuf  = (__bf16*)alloc((size_t)B_ * INP * 2);
    __bf16*   h1    = (__bf16*)alloc((size_t)B_ * H_  * 2);
    __bf16*   h2    = (__bf16*)alloc((size_t)B_ * H_  * 2);
    __bf16*   h3    = (__bf16*)alloc((size_t)B_ * H_  * 2);
    float*    c1    = (float*) alloc((size_t)B_ * H_  * 4);
    float*    c2    = (float*) alloc((size_t)B_ * H_  * 4);
    float*    c3    = (float*) alloc((size_t)B_ * H_  * 4);
    float*    gates = (float*) alloc((size_t)B_ * G4H * 4);
    unsigned* bar   = (unsigned*)alloc(256);
    size_t state_words = (off - state_begin) / 4;

    // zero state (h/c/xbuf padding/barrier) every launch -> deterministic
    {
        int blocks = (int)((state_words + NTHR - 1) / NTHR);
        zero_ws_kernel<<<blocks, NTHR, 0, stream>>>((unsigned*)(ws + state_begin), state_words);
    }

    // fp32 -> bf16 weight conversion (with padding), every launch
    auto cvt = [&](const float* src, __bf16* dst, int dr, int dc, int sr, int sc) {
        int total = dr * dc;
        int blocks = (total + NTHR - 1) / NTHR;
        cvt_pad_kernel<<<blocks, NTHR, 0, stream>>>(src, dst, dr, dc, sr, sc);
    };
    cvt(fWih1, wWih1, G4H, INP, G4H, IN_);
    cvt(fWhh1, wWhh1, G4H, H_,  G4H, H_);
    cvt(fWih2, wWih2, G4H, H_,  G4H, H_);
    cvt(fWhh2, wWhh2, G4H, H_,  G4H, H_);
    cvt(fWih3, wWih3, G4H, H_,  G4H, H_);
    cvt(fWhh3, wWhh3, G4H, H_,  G4H, H_);
    cvt(fWdec, wWdec, OUTP, H_, OUT_, H_);

    lstm_persistent<<<NWG, NTHR, 0, stream>>>(
        seed,
        wWih1, wWhh1, bih1, bhh1,
        wWih2, wWhh2, bih2, bhh2,
        wWih3, wWhh3, bih3, bhh3,
        wWdec, bdec,
        xbuf, h1, h2, h3, c1, c2, c3,
        gates, out, bar);
}
